// LearnedGraphLaplacian_57964878627356
// MI455X (gfx1250) — matrix operand order repaired and checked by reading, not compiled, not run on hardware
//
#include <hip/hip_runtime.h>

typedef _Float16 v16h __attribute__((ext_vector_type(16)));
typedef _Float16 v8h  __attribute__((ext_vector_type(8)));
typedef _Float16 v4h  __attribute__((ext_vector_type(4)));
typedef float    v8f  __attribute__((ext_vector_type(8)));
typedef float    v4f  __attribute__((ext_vector_type(4)));

#define B_   8
#define C_   128
#define CH_  64
#define H_   256
#define W_   256
#define HW_  (H_ * W_)
#define ASTR 136   // f16 stride of sA rows: 272B, 16B-aligned, bank-spread
#define HSTR 68    // f32 stride of sH rows (same bytes as ASTR f16)

__device__ __forceinline__ float softplus_f(float x) {
    return fmaxf(x, 0.0f) + log1pf(expf(-fabsf(x)));
}

// ---------------------------------------------------------------------------
// Kernel A: per-pixel 128->64 GEMM (f16 WMMA, f32 acc) + relu, then 64->4
// GEMM (VALU) + softplus. One block = one (b,h) row of 256 pixels.
// A is staged global->LDS with coalesced B128 loads + f32->f16 transpose-pack,
// then fed to WMMA with two ds_load_b128 per fragment.
// Gate field written as (B,H,W,4) float4 per pixel.
// ---------------------------------------------------------------------------
__global__ __launch_bounds__(256)
void lgl_gate_gemm(const float* __restrict__ win,
                   const float* __restrict__ w1,
                   const float* __restrict__ b1,
                   const float* __restrict__ w2,
                   const float* __restrict__ b2,
                   float* __restrict__ wgate)
{
    // sA (f16 [256][ASTR]) and sH (f32 [256][HSTR]) alias: never live together
    __shared__ __align__(16) unsigned char smemAH[256 * ASTR * 2]; // 69.6 KB
    _Float16* sA = (_Float16*)smemAH;
    float*    sH = (float*)smemAH;

    __shared__ _Float16 sW1[CH_ * C_];        // 16 KB, [o][c] row-major, f16
    __shared__ float    sB1[CH_];
    __shared__ float    sW2[4 * CH_];
    __shared__ float    sB2[4];

    const int tid = threadIdx.x;
    const int row = blockIdx.x;               // b*H + h
    const int b   = row >> 8;
    const int h   = row & (H_ - 1);

    // ---- parameter staging (f32 -> f16 for w1) ----
    #pragma unroll 4
    for (int i = tid; i < CH_ * C_; i += 256) sW1[i] = (_Float16)w1[i];
    if (tid < CH_) sB1[tid] = b1[tid];
    sW2[tid] = w2[tid];                       // exactly 256 elements
    if (tid < 4) sB2[tid] = b2[tid];

    // ---- stage A tile: 128 ch x 256 px f32 -> f16 sA[pixel][channel] ----
    const float* Abase = win + (size_t)b * ((size_t)C_ * HW_) + (size_t)h * W_;
    {
        const int pg = tid & 63;              // pixels 4*pg .. 4*pg+3
        const int cq = tid >> 6;              // 0..3
        #pragma unroll
        for (int i = 0; i < 8; ++i) {
            const int c0 = 16 * i + 4 * cq;   // 4 consecutive channels
            const float* g0 = Abase + (size_t)c0 * HW_ + 4 * pg;
            v4f r0 = *(const v4f*)(g0);
            v4f r1 = *(const v4f*)(g0 + (size_t)HW_);
            v4f r2 = *(const v4f*)(g0 + (size_t)2 * HW_);
            v4f r3 = *(const v4f*)(g0 + (size_t)3 * HW_);
            #pragma unroll
            for (int j = 0; j < 4; ++j) {     // transpose-pack: 4ch per pixel
                v4h pk = { (_Float16)r0[j], (_Float16)r1[j],
                           (_Float16)r2[j], (_Float16)r3[j] };
                *(v4h*)&sA[(size_t)(4 * pg + j) * ASTR + c0] = pk;
            }
        }
    }
    __syncthreads();

    const int wv    = tid >> 5;               // wave id 0..7
    const int lane  = tid & 31;
    const int l16   = lane >> 4;              // 0 = lanes 0-15, 1 = lanes 16-31
    const int lm    = lane & 15;
    const int pbase = wv * 32;                // wave's first pixel in row

    v8f acc[2][4] = {};                       // 2 M-tiles x 4 N-tiles, f32

    #pragma unroll
    for (int ks = 0; ks < 4; ++ks) {          // K = 128 in 4 steps of 32
        const int k0 = ks * 32;

        // --- B fragments (K x N = 32x16 f16): k = (e&7) + 8*l16 + 16*(e>>3), n = nt*16+lm
        v16h bf[4];
        #pragma unroll
        for (int nt = 0; nt < 4; ++nt) {
            const int n = nt * 16 + lm;
            #pragma unroll
            for (int e = 0; e < 16; ++e)
                bf[nt][e] = sW1[n * C_ + k0 + (e & 7) + 8 * l16 + 16 * (e >> 3)];
        }

        // --- A fragments (M x K = 16x32 f16): two ds_load_b128 per fragment
        v16h af[2];
        #pragma unroll
        for (int t = 0; t < 2; ++t) {
            const int p = pbase + t * 16 + lm;    // pixel (row M index)
            const _Float16* fp = &sA[(size_t)p * ASTR + k0 + 8 * l16];
            v8h lo = *(const v8h*)fp;             // k = k0+8*l16 .. +7
            v8h hi = *(const v8h*)(fp + 16);      // k = k0+16+8*l16 .. +7
            af[t] = __builtin_shufflevector(lo, hi,
                        0,1,2,3,4,5,6,7,8,9,10,11,12,13,14,15);
        }

        // --- 8 WMMAs per K-step
        #pragma unroll
        for (int t = 0; t < 2; ++t)
            #pragma unroll
            for (int nt = 0; nt < 4; ++nt)
                acc[t][nt] = __builtin_amdgcn_wmma_f32_16x16x32_f16(
                    false, af[t], false, bf[nt],
                    (short)0, acc[t][nt], false, false);
    }

    // all waves done reading sA before it is reused as sH
    __syncthreads();

    // spill relu(h + b1) to LDS per the 16x16 f32 C/D layout:
    // vgpr r -> M = r + 8*l16, lane%16 -> N
    #pragma unroll
    for (int nt = 0; nt < 4; ++nt) {
        const int o = nt * 16 + lm;
        const float bias = sB1[o];
        #pragma unroll
        for (int t = 0; t < 2; ++t) {
            #pragma unroll
            for (int r = 0; r < 8; ++r) {
                const int p = pbase + t * 16 + r + 8 * l16;
                sH[(size_t)p * HSTR + o] = fmaxf(acc[t][nt][r] + bias, 0.0f);
            }
        }
    }
    __syncthreads();

    // second GEMM: 64 -> 4 per pixel (one thread = one pixel), then softplus
    float a0 = sB2[0], a1 = sB2[1], a2 = sB2[2], a3 = sB2[3];
    const float* hp = &sH[(size_t)tid * HSTR];
    #pragma unroll 8
    for (int o = 0; o < CH_; ++o) {
        const float hv = hp[o];
        a0 = fmaf(hv, sW2[0 * CH_ + o], a0);
        a1 = fmaf(hv, sW2[1 * CH_ + o], a1);
        a2 = fmaf(hv, sW2[2 * CH_ + o], a2);
        a3 = fmaf(hv, sW2[3 * CH_ + o], a3);
    }
    float4 g = make_float4(softplus_f(a0), softplus_f(a1),
                           softplus_f(a2), softplus_f(a3));
    ((float4*)wgate)[(size_t)row * W_ + tid] = g;   // (B,H,W,4)
}

// ---------------------------------------------------------------------------
// Kernel B: gated 5-point Laplacian. One block = one (b,c) plane; threads are
// columns; rolling prev/cur/next row registers -> each u element loaded once.
// ---------------------------------------------------------------------------
__global__ __launch_bounds__(256)
void lgl_stencil(const float* __restrict__ u,
                 const float* __restrict__ wgate,
                 float* __restrict__ out)
{
    const int t  = threadIdx.x;               // column index 0..255
    const int bc = blockIdx.x;                // b*C + c
    const int b  = bc >> 7;

    const size_t plane = (size_t)bc * HW_;
    const float* up = u + plane;
    float*       op = out + plane;
    const float4* wq = (const float4*)wgate + (size_t)b * HW_;

    float uPrev = 0.0f;                       // shift_up pads row -1 with 0
    float uCur  = up[t];

    for (int h = 0; h < H_; ++h) {
        const int base = h * W_;
        if (h < H_ - 2)                       // global_prefetch_b8, 2 rows ahead
            __builtin_prefetch(up + base + 2 * W_ + t, 0, 1);

        const float uNext = (h < H_ - 1) ? up[base + W_ + t] : 0.0f;
        const float uL    = (t > 0)      ? up[base + t - 1]  : 0.0f;
        const float uR    = (t < W_ - 1) ? up[base + t + 1]  : 0.0f;

        const float4 g = wq[(size_t)base + t];

        float lap = g.x * (uPrev - uCur)      // up:    u[h-1] - u[h]
                  + g.y * (uNext - uCur)      // down:  u[h+1] - u[h]
                  + g.z * (uL    - uCur)      // left:  u[w-1] - u[w]
                  + g.w * (uR    - uCur);     // right: u[w+1] - u[w]

        op[base + t] = lap;
        uPrev = uCur;
        uCur  = uNext;
    }
}

// ---------------------------------------------------------------------------
extern "C" void kernel_launch(void* const* d_in, const int* in_sizes, int n_in,
                              void* d_out, int out_size, void* d_ws, size_t ws_size,
                              hipStream_t stream)
{
    const float* u   = (const float*)d_in[0];
    const float* win = (const float*)d_in[1];
    const float* w1  = (const float*)d_in[2];
    const float* b1  = (const float*)d_in[3];
    const float* w2  = (const float*)d_in[4];
    const float* b2  = (const float*)d_in[5];

    float* wgate = (float*)d_ws;              // (B,H,W,4) f32 = 8 MB scratch

    lgl_gate_gemm<<<B_ * H_, 256, 0, stream>>>(win, w1, b1, w2, b2, wgate);
    lgl_stencil <<<B_ * C_, 256, 0, stream>>>(u, wgate, (float*)d_out);
}
